// ShallowSIRENWithPosEmb_60816736911655
// MI455X (gfx1250) — matrix operand order repaired and checked by reading, not compiled, not run on hardware
//
#include <hip/hip_runtime.h>
#include <hip/hip_bf16.h>

typedef __attribute__((ext_vector_type(16))) _Float16 v16h;
typedef __attribute__((ext_vector_type(8)))  float    v8f;

#define SIREN_SCALE (2.0f / 0.24f)

// ---------- fragment index helpers (CDNA5 16-bit WMMA layouts) ----------
// A-matrix 16x32 f16 (ISA 7.12.2): lane = m + g*16 (g = k-group), slot s:
//   k = ((s>>1)&3)*2 + (s>>3)*16 + g*8 + (s&1)
// Inverse (scatter an element (m,k) into the 512-half tile):
__device__ __forceinline__ int a_idx(int m, int k) {
    int g = (k >> 3) & 1;
    int h = k & 1;
    int r = k - (g << 3) - h;
    int v = ((r >> 4) << 2) + ((r & 15) >> 1);
    return (m + (g << 4)) * 16 + v * 2 + h;
}

// B-matrix 32x16 f16 (same k order as A): lane group g holds
// K = g*8 + (slot&7) + (slot>>3)*16, column n = lane&15.  Fragments stored [lane][slot] so each
// lane loads 32 contiguous bytes.
__device__ __forceinline__ v16h loadB(const _Float16* wp, int frag, int lane) {
    return *(const v16h*)(wp + frag * 512 + lane * 16);
}
__device__ __forceinline__ v16h loadA(const _Float16* buf, int kt, int mt, int lane) {
    return *(const v16h*)(buf + (kt * 4 + mt) * 512 + lane * 16);
}
__device__ __forceinline__ v8f wmma16(v16h a, v16h b, v8f c) {
    return __builtin_amdgcn_wmma_f32_16x16x32_f16(false, a, false, b, (short)0, c, false, false);
}

// D layout f32 16x16: lane<16 -> n=lane, m=v ; lane>=16 -> n=lane-16, m=v+8
__device__ __forceinline__ void act_store(v8f acc, _Float16* dst, int mt, int nt, int lane,
                                          const float* styleF, const float* styleP,
                                          const float* bias, int koff) {
    int n = lane & 15;
    int f = nt * 16 + n;
    float fr = styleF[f] * 15.0f + 30.0f;
    float ph = styleP[f];
    float bs = bias[f];
    int kd = koff + f;
    int kt = kd >> 5, ko = kd & 31;
    int mbase = (lane >> 4) << 3;
    _Float16* tile = dst + (kt * 4 + mt) * 512;
#pragma unroll
    for (int i = 0; i < 8; ++i) {
        float h = acc[i] + bs;
        tile[a_idx(mbase + i, ko)] = (_Float16)__sinf(fr * h + ph);
    }
}

template <int KT>
__device__ __forceinline__ void film_layer(const _Float16* inBuf, const _Float16* wfrag,
                                           _Float16* outBuf, int koff, int wave, int lane,
                                           const float* styleF, const float* styleP,
                                           const float* bias) {
#pragma unroll
    for (int half = 0; half < 2; ++half) {
        int nt = wave * 2 + half;
        v8f a0 = {}, a1 = {}, a2 = {}, a3 = {};
#pragma unroll
        for (int kt = 0; kt < KT; ++kt) {
            v16h b = loadB(wfrag, kt * 16 + nt, lane);
            a0 = wmma16(loadA(inBuf, kt, 0, lane), b, a0);
            a1 = wmma16(loadA(inBuf, kt, 1, lane), b, a1);
            a2 = wmma16(loadA(inBuf, kt, 2, lane), b, a2);
            a3 = wmma16(loadA(inBuf, kt, 3, lane), b, a3);
        }
        act_store(a0, outBuf, 0, nt, lane, styleF, styleP, bias, koff);
        act_store(a1, outBuf, 1, nt, lane, styleF, styleP, bias, koff);
        act_store(a2, outBuf, 2, nt, lane, styleF, styleP, bias, koff);
        act_store(a3, outBuf, 3, nt, lane, styleF, styleP, bias, koff);
    }
}

// ---------- weight packing: f32 -> f16 WMMA B-fragments ----------
// frag map: [0,32)  W1  (2 kt x 16 nt, K 63->64 pad)
//           [32,160) W2 (8 kt x 16 nt)
//           [160,304) Wcs (9 kt x 16 nt, K: de 0..26, pad 27..31, x 32..287)
//           [304,320) Wlast (16 kt x 1 nt): n<3 rgb rows (k<256 from Wcl),
//                                           n==3 sigma (k>=256 from Wf)
#define PACK_TOTAL (320 * 512)
__global__ void pack_weights(const float* __restrict__ W1, const float* __restrict__ W2,
                             const float* __restrict__ Wcs, const float* __restrict__ Wf,
                             const float* __restrict__ Wcl, _Float16* __restrict__ wp) {
    int idx = blockIdx.x * blockDim.x + threadIdx.x;
    if (idx >= PACK_TOTAL) return;
    int lane = (idx >> 4) & 31;
    int slot = idx & 15;
    int frag = idx >> 9;
    int n = lane & 15;
    int k = ((lane >> 4) << 3) + (slot & 7) + ((slot >> 3) << 4);
    float val = 0.0f;
    if (frag < 32) {
        int kt = frag >> 4, nt = frag & 15;
        int kg = kt * 32 + k;
        if (kg < 63) val = W1[(nt * 16 + n) * 63 + kg];
    } else if (frag < 160) {
        int f = frag - 32;
        int kt = f >> 4, nt = f & 15;
        val = W2[(nt * 16 + n) * 256 + kt * 32 + k];
    } else if (frag < 304) {
        int f = frag - 160;
        int kt = f >> 4, nt = f & 15;
        int kg = kt * 32 + k;
        if (kg < 27) val = Wcs[(nt * 16 + n) * 283 + kg];
        else if (kg >= 32) val = Wcs[(nt * 16 + n) * 283 + (kg - 5)];
    } else {
        int kt = frag - 304;
        int kg = kt * 32 + k;
        if (n < 3) { if (kg < 256) val = Wcl[n * 256 + kg]; }
        else if (n == 3) { if (kg >= 256) val = Wf[kg - 256]; }
    }
    wp[idx] = (_Float16)val;
}

// ---------- fused SIREN kernel: 64 points per 256-thread block ----------
__global__ __launch_bounds__(256, 1) void siren_fused(
    const float* __restrict__ input, const float* __restrict__ raydir,
    const float* __restrict__ style_net, const float* __restrict__ style_col,
    const float* __restrict__ b1, const float* __restrict__ b2,
    const float* __restrict__ bfv, const float* __restrict__ bcs,
    const float* __restrict__ bcl,
    const _Float16* __restrict__ wpack, float* __restrict__ out) {
    __shared__ __align__(32) _Float16 sPE[2 * 4 * 512];   // posembed 64 x 64
    __shared__ __align__(32) _Float16 sX1[8 * 4 * 512];   // x1, later hc
    __shared__ __align__(32) _Float16 sL3[9 * 4 * 512];   // [de | x2] 64 x 288
    __shared__ float sCoord[192];
    __shared__ float sDir[192];

    int tid = threadIdx.x;
    int lane = tid & 31, wave = tid >> 5;
    long pb = (long)blockIdx.x * 64;
    int b = (int)(pb >> 16);  // 65536 points per batch

    if (tid < 192) {
        sCoord[tid] = input[pb * 3 + tid];
        sDir[tid] = raydir[pb * 3 + tid];
    }
    __syncthreads();

    // positional embedding (K padded 63->64) into sPE
    for (int e = tid; e < 64 * 64; e += 256) {
        int m = e >> 6, k = e & 63;
        float val = 0.0f;
        if (k < 3) {
            val = sCoord[m * 3 + k] * SIREN_SCALE;
        } else if (k < 63) {
            int j = k - 3, fi = j / 6, rem = j % 6;
            int s = rem / 3, c = rem % 3;
            float xb = sCoord[m * 3 + c] * SIREN_SCALE * (float)(1 << fi);
            val = s ? __cosf(xb) : __sinf(xb);
        }
        sPE[((k >> 5) * 4 + (m >> 4)) * 512 + a_idx(m & 15, k & 31)] = (_Float16)val;
    }
    // direction embedding into k-tile 0 of sL3 (k 27..31 zero pad)
    for (int e = tid; e < 64 * 32; e += 256) {
        int m = e >> 5, k = e & 31;
        float val = 0.0f;
        if (k < 3) {
            val = sDir[m * 3 + k];
        } else if (k < 27) {
            int j = k - 3, fi = j / 6, rem = j % 6;
            int s = rem / 3, c = rem % 3;
            float xb = sDir[m * 3 + c] * (float)(1 << fi);
            val = s ? __cosf(xb) : __sinf(xb);
        }
        sL3[(m >> 4) * 512 + a_idx(m & 15, k)] = (_Float16)val;
    }
    __syncthreads();

    const _Float16* W1p = wpack;
    const _Float16* W2p = wpack + 32 * 512;
    const _Float16* Wcsp = wpack + 160 * 512;
    const _Float16* Wlp = wpack + 304 * 512;
    const float* sn = style_net + (long)b * 1024;
    const float* sc = style_col + (long)b * 512;

    film_layer<2>(sPE, W1p, sX1, 0, wave, lane, sn, sn + 512, b1);        // x1
    __syncthreads();
    film_layer<8>(sX1, W2p, sL3, 32, wave, lane, sn + 256, sn + 768, b2); // x2 -> sL3 kt1..8
    __syncthreads();
    film_layer<9>(sL3, Wcsp, sX1, 0, wave, lane, sc, sc + 256, bcs);      // hc -> sX1
    __syncthreads();

    // final heads as one 512x16 WMMA chain: n<3 rgb, n==3 sigma
    if (wave < 4) {
        int mt = wave;
        v8f acc = {};
#pragma unroll
        for (int kt = 0; kt < 16; ++kt) {
            v16h a = (kt < 8) ? loadA(sX1, kt, mt, lane) : loadA(sL3, kt - 7, mt, lane);
            acc = wmma16(a, loadB(Wlp, kt, lane), acc);
        }
        int n = lane & 15;
        if (n < 4) {
            float bias = (n < 3) ? bcl[n] : bfv[0];
            int mbase = (lane >> 4) << 3;
#pragma unroll
            for (int i = 0; i < 8; ++i) {
                long p = pb + mt * 16 + mbase + i;
                out[p * 4 + n] = acc[i] + bias;
            }
        }
    }
}

extern "C" void kernel_launch(void* const* d_in, const int* in_sizes, int n_in,
                              void* d_out, int out_size, void* d_ws, size_t ws_size,
                              hipStream_t stream) {
    const float* input  = (const float*)d_in[0];
    const float* raydir = (const float*)d_in[1];
    const float* snet   = (const float*)d_in[2];
    const float* scol   = (const float*)d_in[3];
    const float* W1     = (const float*)d_in[4];
    const float* b1     = (const float*)d_in[5];
    const float* W2     = (const float*)d_in[6];
    const float* b2     = (const float*)d_in[7];
    const float* Wf     = (const float*)d_in[8];
    const float* bf     = (const float*)d_in[9];
    const float* Wcs    = (const float*)d_in[10];
    const float* bcs    = (const float*)d_in[11];
    const float* Wcl    = (const float*)d_in[12];
    const float* bcl    = (const float*)d_in[13];
    _Float16* wpack = (_Float16*)d_ws;  // needs 320*512*2 = 320KB of scratch
    float* out = (float*)d_out;

    pack_weights<<<(PACK_TOTAL + 255) / 256, 256, 0, stream>>>(W1, W2, Wcs, Wf, Wcl, wpack);

    int total_points = in_sizes[0] / 3;       // B*N = 262144
    int blocks = total_points / 64;           // 4096
    siren_fused<<<blocks, 256, 0, stream>>>(input, raydir, snet, scol,
                                            b1, b2, bf, bcs, bcl, wpack, out);
}
